// Dynamic_Adaptive_Scan_27771258536646
// MI455X (gfx1250) — matrix operand order repaired and checked by reading, not compiled, not run on hardware
//
#include <hip/hip_runtime.h>
#include <hip/hip_bf16.h>
#include <math.h>

typedef __attribute__((ext_vector_type(16))) _Float16 v16h;
typedef __attribute__((ext_vector_type(8)))  _Float16 v8h;
typedef __attribute__((ext_vector_type(8)))  float    v8f;

#define NN 2
#define CC 128
#define HH 128
#define WW 128
#define GRP 4
#define GC 32
#define PP 9
#define OD 72          // GROUP*P*2
#define ODP 80         // padded to 5 n-tiles
#define MROWS (NN*HH*WW)   // 32768
#define MTILES (MROWS/16)  // 2048
#define NTILES (ODP/16)    // 5
#define KTILES (CC/32)     // 4

// ---------------------------------------------------------------------------
// Kernel A: depthwise 3x3 conv (NCHW in) + bias + LayerNorm over C + GELU,
// writes x1 as f16 NHWC for the WMMA GEMM.
// grid: N*H*2 blocks (each block: 64 w positions x 128 channels), 256 threads
// ---------------------------------------------------------------------------
__global__ void dw_ln_gelu_kernel(const float* __restrict__ inp,
                                  const float* __restrict__ dw_w,
                                  const float* __restrict__ dw_b,
                                  const float* __restrict__ ln_g,
                                  const float* __restrict__ ln_b,
                                  _Float16* __restrict__ x1h) {
  __shared__ float sw[9 * CC];
  __shared__ float sb[CC];
  __shared__ float sy[CC * 64];      // [c][w] conv result, 32 KB
  __shared__ float red_s[64][4];
  __shared__ float red_q[64][4];
  __shared__ float s_mu[64];
  __shared__ float s_rs[64];

  const int bid = blockIdx.x;
  const int wh  = bid & 1;
  const int h   = (bid >> 1) & (HH - 1);
  const int n   = bid >> 8;
  const int w0  = wh * 64;
  const int tid = threadIdx.x;

  for (int i = tid; i < 9 * CC; i += 256) sw[i] = dw_w[i];
  for (int i = tid; i < CC; i += 256)     sb[i] = dw_b[i];
  __syncthreads();

  // phase 1: depthwise conv, coalesced along w
  const int wl = tid & 31;   // w lane
  const int cg = tid >> 5;   // channel group 0..7
  for (int ci = 0; ci < 16; ++ci) {
    const int c = cg * 16 + ci;
    const float* ibase = inp + (((size_t)n * CC + c) * HH) * WW;
    for (int wi = 0; wi < 2; ++wi) {
      const int w = w0 + wl + wi * 32;
      float s = sb[c];
      #pragma unroll
      for (int ky = 0; ky < 3; ++ky) {
        const int y = h + ky - 1;
        if (y < 0 || y >= HH) continue;
        #pragma unroll
        for (int kx = 0; kx < 3; ++kx) {
          const int x = w + kx - 1;
          if (x < 0 || x >= WW) continue;
          s += ibase[y * WW + x] * sw[(ky * 3 + kx) * CC + c];
        }
      }
      sy[c * 64 + (w - w0)] = s;
    }
  }
  __syncthreads();

  // phase 2: LayerNorm over C per pixel (4 threads/pixel), then GELU
  const int px = tid >> 2;   // pixel 0..63
  const int q  = tid & 3;    // quarter of channels
  float sum = 0.f, sq = 0.f;
  for (int i = 0; i < 32; ++i) {
    const float v = sy[(q * 32 + i) * 64 + px];
    sum += v; sq += v * v;
  }
  red_s[px][q] = sum;
  red_q[px][q] = sq;
  __syncthreads();
  if (q == 0) {
    const float S = red_s[px][0] + red_s[px][1] + red_s[px][2] + red_s[px][3];
    const float Q = red_q[px][0] + red_q[px][1] + red_q[px][2] + red_q[px][3];
    const float mu  = S * (1.f / CC);
    const float var = Q * (1.f / CC) - mu * mu;
    s_mu[px] = mu;
    s_rs[px] = rsqrtf(var + 1e-6f);
  }
  __syncthreads();
  const float mu = s_mu[px];
  const float rs = s_rs[px];
  const int w = w0 + px;
  _Float16* orow = x1h + ((size_t)((n * HH + h) * WW + w)) * CC;
  for (int i = 0; i < 32; ++i) {
    const int c = q * 32 + i;
    const float v = (sy[c * 64 + px] - mu) * rs * ln_g[c] + ln_b[c];
    const float g = 0.5f * v * (1.f + erff(v * 0.70710678118654752f));
    orow[c] = (_Float16)g;
  }
}

// ---------------------------------------------------------------------------
// Kernel P: pack off_w [72][128] f32 into per-wave WMMA B fragments, f16.
// Layout: [ktile(4)][ntile(5)][lane(32)][elem(16)]; B lane mapping per ISA:
//   n = lane%16, k = ktile*32 + (elem/8)*16 + (lane/16)*8 + elem%8
// ---------------------------------------------------------------------------
__global__ void pack_b_kernel(const float* __restrict__ off_w,
                              _Float16* __restrict__ bpack) {
  const int idx = blockIdx.x * 256 + threadIdx.x;
  if (idx >= KTILES * NTILES * 32 * 16) return;
  const int e    = idx & 15;
  const int lane = (idx >> 4) & 31;
  const int nt   = (idx >> 9) % NTILES;
  const int kt   = idx / (NTILES * 512);
  const int k = kt * 32 + ((e >> 3) * 16) + ((lane >> 4) * 8) + (e & 7);
  const int o = nt * 16 + (lane & 15);
  bpack[idx] = (_Float16)((o < OD) ? off_w[o * CC + k] : 0.f);
}

// ---------------------------------------------------------------------------
// Kernel B: offset = x1h[32768x128] * off_w^T + off_b.
// One wave per 16-row M strip; the wave keeps the full padded N (5 tiles) in
// registers, so each A fragment load feeds 5 WMMAs (A traffic cut 5x).
// 20 x v_wmma_f32_16x16x32_f16 per wave.
// grid: 2048 waves / 8 per block = 256 blocks of 256 threads
// ---------------------------------------------------------------------------
__device__ __forceinline__ void store_strip(float* __restrict__ offset,
                                            const float* __restrict__ off_b,
                                            int rbase, int col, const v8f& acc) {
  if (col < OD) {
    const float bias = off_b[col];
    #pragma unroll
    for (int v = 0; v < 8; ++v)
      offset[(size_t)(rbase + v) * OD + col] = acc[v] + bias;
  }
}

__global__ void offset_gemm_kernel(const _Float16* __restrict__ x1h,
                                   const _Float16* __restrict__ bpack,
                                   const float* __restrict__ off_b,
                                   float* __restrict__ offset) {
  const int mt   = (blockIdx.x * blockDim.x + threadIdx.x) >> 5;  // M tile id
  const int lane = threadIdx.x & 31;
  if (mt >= MTILES) return;            // wave-uniform (EXEC stays all-ones)

  const int mrow  = mt * 16 + (lane & 15);
  const int khalf = (lane >> 4) * 8;   // A lane mapping: k = kt*32 + {0|16} + khalf + e
  const _Float16* arow  = x1h + (size_t)mrow * CC + khalf;
  const _Float16* bbase = bpack + (size_t)lane * 16;

  v8f acc0 = {}, acc1 = {}, acc2 = {}, acc3 = {}, acc4 = {};
  #pragma unroll
  for (int kt = 0; kt < KTILES; ++kt) {
    if (kt < KTILES - 1)
      __builtin_prefetch(arow + (kt + 1) * 32, 0, 0);   // global_prefetch_b8
    const v8h alo = *(const v8h*)(arow + kt * 32);
    const v8h ahi = *(const v8h*)(arow + kt * 32 + 16);
    v16h a;
    #pragma unroll
    for (int e = 0; e < 8; ++e) { a[e] = alo[e]; a[e + 8] = ahi[e]; }

    const _Float16* bk = bbase + (size_t)kt * NTILES * 512;   // 512 halves per n-tile
    v16h b;
    b = *(const v16h*)(bk + 0 * 512);
    acc0 = __builtin_amdgcn_wmma_f32_16x16x32_f16(false, a, false, b, (short)0, acc0, false, false);
    b = *(const v16h*)(bk + 1 * 512);
    acc1 = __builtin_amdgcn_wmma_f32_16x16x32_f16(false, a, false, b, (short)0, acc1, false, false);
    b = *(const v16h*)(bk + 2 * 512);
    acc2 = __builtin_amdgcn_wmma_f32_16x16x32_f16(false, a, false, b, (short)0, acc2, false, false);
    b = *(const v16h*)(bk + 3 * 512);
    acc3 = __builtin_amdgcn_wmma_f32_16x16x32_f16(false, a, false, b, (short)0, acc3, false, false);
    b = *(const v16h*)(bk + 4 * 512);
    acc4 = __builtin_amdgcn_wmma_f32_16x16x32_f16(false, a, false, b, (short)0, acc4, false, false);
  }

  // D layout: VGPR v -> row = v + (lane/16)*8, col = lane%16
  const int rbase = mt * 16 + (lane >> 4) * 8;
  const int nl = lane & 15;
  store_strip(offset, off_b, rbase, 0 * 16 + nl, acc0);
  store_strip(offset, off_b, rbase, 1 * 16 + nl, acc1);
  store_strip(offset, off_b, rbase, 2 * 16 + nl, acc2);
  store_strip(offset, off_b, rbase, 3 * 16 + nl, acc3);
  store_strip(offset, off_b, rbase, 4 * 16 + nl, acc4);
}

// ---------------------------------------------------------------------------
// Kernel C: DCNv3 sampling. One wave per (pixel, group); lane = channel in
// group. Padded coords: ix = w + i + off_x, iy = h + j + off_y; sample x iff
// coord in [1,128]. Stage in LDS -> coalesced NCHW store.
// grid: N*H*2 blocks (64 w each), 256 threads = 8 waves
// ---------------------------------------------------------------------------
__device__ __forceinline__ float dcn_sample(const float* __restrict__ xb,
                                            int yy, int xx) {
  if (yy < 1 || yy > HH || xx < 1 || xx > WW) return 0.f;
  return xb[((size_t)(yy - 1) * WW + (xx - 1)) * CC];
}

__global__ void dcnv3_kernel(const float* __restrict__ x,
                             const float* __restrict__ offset,
                             float* __restrict__ out) {
  __shared__ float acc_s[CC * 65];   // [c][w] padded stride, 33.3 KB

  const int bid = blockIdx.x;
  const int wh  = bid & 1;
  const int h   = (bid >> 1) & (HH - 1);
  const int n   = bid >> 8;
  const int w0  = wh * 64;
  const int tid = threadIdx.x;
  const int waveId = tid >> 5;
  const int lane   = tid & 31;

  for (int t = waveId; t < 256; t += 8) {
    const int g   = t >> 6;      // 0..3
    const int pix = t & 63;
    const int w   = w0 + pix;
    const float* offp = offset + ((size_t)((n * HH + h) * WW + w)) * OD + g * (PP * 2);
    const int c = g * GC + lane;
    const float* xb = x + ((size_t)n * HH * WW) * CC + c;
    float acc = 0.f;
    #pragma unroll
    for (int i = 0; i < 3; ++i) {
      #pragma unroll
      for (int j = 0; j < 3; ++j) {
        const int p = i * 3 + j;
        const float ox = offp[p * 2 + 0];
        const float oy = offp[p * 2 + 1];
        const float ix = (float)(w + i) + ox;
        const float iy = (float)(h + j) + oy;
        const float x0f = floorf(ix), y0f = floorf(iy);
        const float wx = ix - x0f, wy = iy - y0f;
        const int x0 = (int)x0f, y0 = (int)y0f;
        acc += dcn_sample(xb, y0,     x0    ) * (1.f - wy) * (1.f - wx);
        acc += dcn_sample(xb, y0,     x0 + 1) * (1.f - wy) * wx;
        acc += dcn_sample(xb, y0 + 1, x0    ) * wy * (1.f - wx);
        acc += dcn_sample(xb, y0 + 1, x0 + 1) * wy * wx;
      }
    }
    acc_s[c * 65 + pix] = acc;
  }
  __syncthreads();

  for (int idx = tid; idx < CC * 64; idx += 256) {
    const int c = idx >> 6;
    const int w = idx & 63;
    out[(((size_t)(n * CC + c) * HH) + h) * WW + w0 + w] = acc_s[c * 65 + w];
  }
}

// ---------------------------------------------------------------------------
extern "C" void kernel_launch(void* const* d_in, const int* in_sizes, int n_in,
                              void* d_out, int out_size, void* d_ws, size_t ws_size,
                              hipStream_t stream) {
  const float* inp   = (const float*)d_in[0];   // [N,C,H,W]
  const float* x     = (const float*)d_in[1];   // [N,H,W,C]
  const float* dw_w  = (const float*)d_in[2];   // [3,3,1,C]
  const float* dw_b  = (const float*)d_in[3];   // [C]
  const float* ln_g  = (const float*)d_in[4];   // [C]
  const float* ln_b  = (const float*)d_in[5];   // [C]
  const float* off_w = (const float*)d_in[6];   // [72,C]
  const float* off_b = (const float*)d_in[7];   // [72]
  float* out = (float*)d_out;                   // [N,C,H,W]

  // workspace layout
  char* ws = (char*)d_ws;
  float*    offset = (float*)ws;                                    // 32768*72*4 = 9437184 B
  _Float16* x1h    = (_Float16*)(ws + (size_t)MROWS * OD * 4);      // 32768*128*2 = 8388608 B
  _Float16* bpack  = (_Float16*)(ws + (size_t)MROWS * OD * 4
                                    + (size_t)MROWS * CC * 2);      // 20480 B

  // 1) depthwise conv + LN + GELU -> x1 (f16)
  dw_ln_gelu_kernel<<<NN * HH * 2, 256, 0, stream>>>(inp, dw_w, dw_b, ln_g, ln_b, x1h);

  // 2a) pack B fragments for WMMA
  pack_b_kernel<<<(KTILES * NTILES * 32 * 16 + 255) / 256, 256, 0, stream>>>(off_w, bpack);

  // 2b) offset GEMM via v_wmma_f32_16x16x32_f16 (20 WMMA / wave)
  offset_gemm_kernel<<<MTILES / 8, 256, 0, stream>>>(x1h, bpack, off_b, offset);

  // 3) DCNv3 bilinear gather + NCHW transpose
  dcnv3_kernel<<<NN * HH * 2, 256, 0, stream>>>(x, offset, out);
}
